// Block_54107997995302
// MI455X (gfx1250) — matrix operand order repaired and checked by reading, not compiled, not run on hardware
//
#include <hip/hip_runtime.h>
#include <hip/hip_bf16.h>

typedef __attribute__((ext_vector_type(16))) _Float16 v16h;
typedef __attribute__((ext_vector_type(8)))  _Float16 v8h;
typedef __attribute__((ext_vector_type(8)))  float    v8f;

#define TB  2
#define TT  2048
#define TC  1024
#define THh 16
#define THD 64
#define TM  (TB*TT)   // 4096 rows

// ---------------------------------------------------------------------------
// LayerNorm: fp32 [rows, C] -> f16 [rows, C].  One 256-thread block per row.
// ---------------------------------------------------------------------------
__global__ void ln_f16_kernel(const float* __restrict__ x,
                              const float* __restrict__ g,
                              const float* __restrict__ beta,
                              _Float16* __restrict__ out) {
  const int row = blockIdx.x;
  const int tid = threadIdx.x;
  const float* xr = x + (size_t)row * TC;
  __shared__ float red[256];

  float s = 0.f;
  for (int i = tid; i < TC; i += 256) s += xr[i];
  red[tid] = s; __syncthreads();
  for (int off = 128; off > 0; off >>= 1) {
    if (tid < off) red[tid] += red[tid + off];
    __syncthreads();
  }
  const float mu = red[0] * (1.0f / TC);
  __syncthreads();

  float v = 0.f;
  for (int i = tid; i < TC; i += 256) { float d = xr[i] - mu; v += d * d; }
  red[tid] = v; __syncthreads();
  for (int off = 128; off > 0; off >>= 1) {
    if (tid < off) red[tid] += red[tid + off];
    __syncthreads();
  }
  const float rstd = rsqrtf(red[0] * (1.0f / TC) + 1e-5f);

  for (int i = tid; i < TC; i += 256)
    out[(size_t)row * TC + i] = (_Float16)((xr[i] - mu) * rstd * g[i] + beta[i]);
}

// ---------------------------------------------------------------------------
// Weight prep: Wq/Wk/Wv [H,C,HD] fp32 -> f16 W'^T[j, c] with j = h*HD+d
// ---------------------------------------------------------------------------
__global__ void qkv_transpose_kernel(const float* __restrict__ W,
                                     _Float16* __restrict__ WT) {
  const size_t idx = (size_t)blockIdx.x * 256 + threadIdx.x;  // over C*C
  const int j = (int)(idx / TC);
  const int c = (int)(idx % TC);
  const int h = j / THD;
  const int d = j % THD;
  WT[idx] = (_Float16)W[(size_t)h * TC * THD + (size_t)c * THD + d];
}

// Generic: W [K,N] fp32 row-major -> WT [N,K] f16
__global__ void transpose_f16_kernel(const float* __restrict__ W,
                                     _Float16* __restrict__ WT,
                                     int K, int N) {
  const size_t idx = (size_t)blockIdx.x * 256 + threadIdx.x;  // over N*K
  const int n = (int)(idx / K);
  const int k = (int)(idx % K);
  WT[idx] = (_Float16)W[(size_t)k * N + n];
}

// V [B*T, C] f16 (head-packed) -> Vt [B, H, HD, T] f16 (coalesced writes)
__global__ void v_transpose_kernel(const _Float16* __restrict__ Vsrc,
                                   _Float16* __restrict__ Vt) {
  const size_t idx = (size_t)blockIdx.x * 256 + threadIdx.x;  // over B*H*HD*T
  const int s    = (int)(idx % TT);
  const size_t r = idx / TT;
  const int d    = (int)(r % THD);
  const int h    = (int)((r / THD) % THh);
  const int b    = (int)(r / (THD * THh));
  Vt[idx] = Vsrc[((size_t)b * TT + s) * TC + h * THD + d];
}

// ---------------------------------------------------------------------------
// WMMA GEMM: Out[M,N] = A[M,K](f16,row) * BT[N,K](f16,row) (+bias)(+relu)(+res)
// One wave computes a 32x64 tile: 2 A-frags x 4 B-frags -> 8 WMMAs / K-step.
// 8 waves / block.
// ---------------------------------------------------------------------------
template <bool OUTF16, bool RELU, bool HASBIAS, bool HASRES>
__global__ void gemm_wmma_kernel(const _Float16* __restrict__ A,
                                 const _Float16* __restrict__ BT,
                                 const float* __restrict__ bias,
                                 const float* __restrict__ res,
                                 void* __restrict__ outp,
                                 int M, int N, int Kd) {
  const int wave = threadIdx.x >> 5;
  const int lane = threadIdx.x & 31;
  const int tilesN = N >> 6;                       // 64-wide tiles
  const long tile = (long)blockIdx.x * 8 + wave;
  const long totalTiles = (long)(M >> 5) * tilesN; // 32-tall tiles
  if (tile >= totalTiles) return;                  // wave-uniform

  const int tm = (int)(tile / tilesN) << 5;
  const int tn = (int)(tile % tilesN) << 6;
  const int nLane = lane & 15;
  const int half  = lane >> 4;

  // A frags: lane holds rows (tm+nLane) and (tm+16+nLane); k chunks half*8, +16
  const _Float16* pa0 = A + (size_t)(tm + nLane) * Kd + half * 8;
  const _Float16* pa1 = A + (size_t)(tm + 16 + nLane) * Kd + half * 8;
  // B frags: lane holds col (tn+j*16+nLane); k chunks half*8, +16 of W^T
  const _Float16* pb[4];
  #pragma unroll
  for (int j = 0; j < 4; ++j)
    pb[j] = BT + (size_t)(tn + j * 16 + nLane) * Kd + half * 8;

  v8f acc[2][4];
  #pragma unroll
  for (int i = 0; i < 2; ++i)
    #pragma unroll
    for (int j = 0; j < 4; ++j)
      #pragma unroll
      for (int r = 0; r < 8; ++r) acc[i][j][r] = 0.f;

  for (int k0 = 0; k0 < Kd; k0 += 32) {
    v8h a0lo = *(const v8h*)(pa0 + k0);
    v8h a0hi = *(const v8h*)(pa0 + k0 + 16);
    v8h a1lo = *(const v8h*)(pa1 + k0);
    v8h a1hi = *(const v8h*)(pa1 + k0 + 16);
    v16h a0 = __builtin_shufflevector(a0lo, a0hi,
        0,1,2,3,4,5,6,7,8,9,10,11,12,13,14,15);
    v16h a1 = __builtin_shufflevector(a1lo, a1hi,
        0,1,2,3,4,5,6,7,8,9,10,11,12,13,14,15);
    v16h b[4];
    #pragma unroll
    for (int j = 0; j < 4; ++j) b[j] = __builtin_shufflevector(*(const v8h*)(pb[j] + k0), *(const v8h*)(pb[j] + k0 + 16), 0,1,2,3,4,5,6,7,8,9,10,11,12,13,14,15);
    #pragma unroll
    for (int j = 0; j < 4; ++j)
      acc[0][j] = __builtin_amdgcn_wmma_f32_16x16x32_f16(
          false, a0, false, b[j], (short)0, acc[0][j], false, false);
    #pragma unroll
    for (int j = 0; j < 4; ++j)
      acc[1][j] = __builtin_amdgcn_wmma_f32_16x16x32_f16(
          false, a1, false, b[j], (short)0, acc[1][j], false, false);
  }

  #pragma unroll
  for (int j = 0; j < 4; ++j) {
    const int n = tn + j * 16 + nLane;
    const float bv = HASBIAS ? bias[n] : 0.f;
    #pragma unroll
    for (int i = 0; i < 2; ++i) {
      #pragma unroll
      for (int r = 0; r < 8; ++r) {
        const int m = tm + i * 16 + half * 8 + r;
        float val = acc[i][j][r] + bv;
        if (RELU) val = fmaxf(val, 0.f);
        if (HASRES) val += res[(size_t)m * N + n];
        if (OUTF16) ((_Float16*)outp)[(size_t)m * N + n] = (_Float16)val;
        else        ((float*)outp)[(size_t)m * N + n] = val;
      }
    }
  }
}

// ---------------------------------------------------------------------------
// Flash attention: one wave per (b, h, 16-query tile), 32 keys per iteration.
// Q/K f16 packed [B*T, C] (head offset h*HD); V pre-transposed [B,H,HD,T] so
// every WMMA fragment (Q, K, P, V) is a contiguous vector load.
// ---------------------------------------------------------------------------
__global__ void attn_kernel(const _Float16* __restrict__ Q,
                            const _Float16* __restrict__ K,
                            const _Float16* __restrict__ Vt,
                            _Float16* __restrict__ O) {
  const int q0 = blockIdx.x * 16;
  const int h  = blockIdx.y;
  const int b  = blockIdx.z;
  const int lane  = threadIdx.x;
  const int nLane = lane & 15;
  const int half  = lane >> 4;
  const size_t base  = ((size_t)b * TT) * TC + (size_t)h * THD;
  const size_t vbase = ((size_t)(b * THh + h)) * THD * TT;   // Vt[b,h,:,:]

  __shared__ float    sS[16 * 32];
  __shared__ _Float16 sP[16 * 32];
  __shared__ float    sM[16], sL[16], sCorr[16];

  if (lane < 16) { sM[lane] = -__builtin_inff(); sL[lane] = 0.f; }
  __syncthreads();

  v8f acc[4];
  #pragma unroll
  for (int i = 0; i < 4; ++i)
    #pragma unroll
    for (int r = 0; r < 8; ++r) acc[i][r] = 0.f;

  const int nT = (q0 + 47) >> 5;          // covers keys 0..q0+15 in 32-chunks
  for (int kt = 0; kt < nT; ++kt) {
    const int s0 = kt * 32;

    // ---- two S tiles: keys [s0, s0+16) and [s0+16, s0+32) ----
    v8f c0 = {}, c1 = {};
    #pragma unroll
    for (int k0 = 0; k0 < THD; k0 += 32) {
      const _Float16* pa = Q + base + (size_t)(q0 + nLane) * TC + k0 + half * 8;
      v8h alo = *(const v8h*)pa;
      v8h ahi = *(const v8h*)(pa + 16);
      v16h a = __builtin_shufflevector(alo, ahi,
          0,1,2,3,4,5,6,7,8,9,10,11,12,13,14,15);
      const _Float16* pk0 = K + base + (size_t)(s0 + nLane) * TC + k0 + half * 8;
      const _Float16* pk1 = K + base + (size_t)(s0 + 16 + nLane) * TC + k0 + half * 8;
      v16h b0 = __builtin_shufflevector(*(const v8h*)pk0, *(const v8h*)(pk0 + 16), 0,1,2,3,4,5,6,7,8,9,10,11,12,13,14,15);
      v16h b1 = __builtin_shufflevector(*(const v8h*)pk1, *(const v8h*)(pk1 + 16), 0,1,2,3,4,5,6,7,8,9,10,11,12,13,14,15);
      c0 = __builtin_amdgcn_wmma_f32_16x16x32_f16(false, a, false, b0,
                                                  (short)0, c0, false, false);
      c1 = __builtin_amdgcn_wmma_f32_16x16x32_f16(false, a, false, b1,
                                                  (short)0, c1, false, false);
    }

    __syncthreads();          // prior-iter sP reads complete
    #pragma unroll
    for (int r = 0; r < 8; ++r) {
      const int mr = r + half * 8;
      const int t  = q0 + mr;
      float v0 = c0[r] * 0.125f;                         // 1/sqrt(64)
      float v1 = c1[r] * 0.125f;
      if (s0 + nLane      > t) v0 = -__builtin_inff();   // causal mask
      if (s0 + 16 + nLane > t) v1 = -__builtin_inff();
      sS[mr * 32 + nLane]      = v0;
      sS[mr * 32 + 16 + nLane] = v1;
    }
    __syncthreads();

    // ---- online softmax: one row (32 scores) per lane 0..15 ----
    if (lane < 16) {
      const int mr = lane;
      const float mOld = sM[mr];
      float mNew = mOld;
      for (int j = 0; j < 32; ++j) mNew = fmaxf(mNew, sS[mr * 32 + j]);
      const float corr = __expf(mOld - mNew);
      float ls = 0.f;
      for (int j = 0; j < 32; ++j) {
        const float p = __expf(sS[mr * 32 + j] - mNew);
        sP[mr * 32 + j] = (_Float16)p;
        ls += p;
      }
      sL[mr] = sL[mr] * corr + ls;
      sM[mr] = mNew;
      sCorr[mr] = corr;
    }
    __syncthreads();

    // ---- rescale accumulator, then acc += P(16x32) * V(32x64) ----
    #pragma unroll
    for (int nc = 0; nc < 4; ++nc)
      #pragma unroll
      for (int r = 0; r < 8; ++r) acc[nc][r] *= sCorr[r + half * 8];

    // P A-frag (full 16x32): elems 0..7 = k half*8.., elems 8..15 = +16
    v8h plo = *(const v8h*)&sP[nLane * 32 + half * 8];
    v8h phi = *(const v8h*)&sP[nLane * 32 + half * 8 + 16];
    v16h pfr = __builtin_shufflevector(plo, phi,
        0,1,2,3,4,5,6,7,8,9,10,11,12,13,14,15);

    // V B-frags: Vt row (d = nc*16+nLane), keys s0+half*8.., +16
    #pragma unroll
    for (int nc = 0; nc < 4; ++nc) {
      const _Float16* pv = Vt + vbase + (size_t)(nc * 16 + nLane) * TT
                              + s0 + half * 8;
      v16h bv = __builtin_shufflevector(*(const v8h*)pv, *(const v8h*)(pv + 16), 0,1,2,3,4,5,6,7,8,9,10,11,12,13,14,15);
      acc[nc] = __builtin_amdgcn_wmma_f32_16x16x32_f16(false, pfr, false, bv,
                                                       (short)0, acc[nc],
                                                       false, false);
    }
  }

  __syncthreads();
  #pragma unroll
  for (int nc = 0; nc < 4; ++nc)
    #pragma unroll
    for (int r = 0; r < 8; ++r) {
      const int mr = r + half * 8;
      const float o = acc[nc][r] / sL[mr];
      O[base + (size_t)(q0 + mr) * TC + nc*16 + nLane] = (_Float16)o;
    }
}

// ---------------------------------------------------------------------------
extern "C" void kernel_launch(void* const* d_in, const int* in_sizes, int n_in,
                              void* d_out, int out_size, void* d_ws, size_t ws_size,
                              hipStream_t stream) {
  const float* x    = (const float*)d_in[0];
  const float* Wq   = (const float*)d_in[1];
  const float* Wk   = (const float*)d_in[2];
  const float* Wv   = (const float*)d_in[3];
  const float* Wp   = (const float*)d_in[4];
  const float* bp   = (const float*)d_in[5];
  const float* W1   = (const float*)d_in[6];
  const float* b1   = (const float*)d_in[7];
  const float* W2   = (const float*)d_in[8];
  const float* b2   = (const float*)d_in[9];
  const float* g1   = (const float*)d_in[10];
  const float* be1  = (const float*)d_in[11];
  const float* g2   = (const float*)d_in[12];
  const float* be2  = (const float*)d_in[13];
  float* out = (float*)d_out;

  // workspace layout (f16 elements); Aelem = 4M, CC = 1M => 52M halfs = 104MB
  _Float16* ws = (_Float16*)d_ws;
  const size_t Aelem = (size_t)TM * TC;     // 4M
  const size_t CC    = (size_t)TC * TC;     // 1M
  _Float16* h1   = ws;                      // LN1 out, later LN2 out
  _Float16* q    = ws + 1 * Aelem;
  _Float16* kb   = ws + 2 * Aelem;
  _Float16* vb   = ws + 3 * Aelem;
  _Float16* attO = ws + 4 * Aelem;
  _Float16* ff1  = ws + 5 * Aelem;          // 4*Aelem
  _Float16* WqT  = ws + 9 * Aelem;
  _Float16* WkT  = WqT + CC;
  _Float16* WvT  = WkT + CC;
  _Float16* WpT  = WvT + CC;
  _Float16* W1T  = WpT + CC;                // [4096,1024] -> 4*CC
  _Float16* W2T  = W1T + 4 * CC;            // [1024,4096] -> 4*CC
  _Float16* Vt   = ws + 12 * Aelem;         // [B,H,HD,T]  -> 1*Aelem

  // ---- weight prep (f16, transposed for contiguous B-fragments) ----
  qkv_transpose_kernel<<<(int)(CC / 256), 256, 0, stream>>>(Wq, WqT);
  qkv_transpose_kernel<<<(int)(CC / 256), 256, 0, stream>>>(Wk, WkT);
  qkv_transpose_kernel<<<(int)(CC / 256), 256, 0, stream>>>(Wv, WvT);
  transpose_f16_kernel<<<(int)(CC / 256), 256, 0, stream>>>(Wp, WpT, TC, TC);
  transpose_f16_kernel<<<(int)(4 * CC / 256), 256, 0, stream>>>(W1, W1T, TC, 4 * TC);
  transpose_f16_kernel<<<(int)(4 * CC / 256), 256, 0, stream>>>(W2, W2T, 4 * TC, TC);

  // ---- LN1 ----
  ln_f16_kernel<<<TM, 256, 0, stream>>>(x, g1, be1, h1);

  // ---- QKV projections: [4096,1024] x [1024,1024], 32x64 tiles ----
  const int blocksCxC  = (TM / 32) * (TC / 64) / 8;       // 256 blocks
  gemm_wmma_kernel<true,  false, false, false><<<blocksCxC, 256, 0, stream>>>(
      h1, WqT, nullptr, nullptr, q,  TM, TC, TC);
  gemm_wmma_kernel<true,  false, false, false><<<blocksCxC, 256, 0, stream>>>(
      h1, WkT, nullptr, nullptr, kb, TM, TC, TC);
  gemm_wmma_kernel<true,  false, false, false><<<blocksCxC, 256, 0, stream>>>(
      h1, WvT, nullptr, nullptr, vb, TM, TC, TC);

  // ---- V head-transpose for contiguous P*V B-fragments ----
  v_transpose_kernel<<<(int)(Aelem / 256), 256, 0, stream>>>(vb, Vt);

  // ---- causal attention ----
  attn_kernel<<<dim3(TT / 16, THh, TB), 32, 0, stream>>>(q, kb, Vt, attO);

  // ---- output projection + residual -> d_out (fp32) ----
  gemm_wmma_kernel<false, false, true, true><<<blocksCxC, 256, 0, stream>>>(
      attO, WpT, bp, x, out, TM, TC, TC);

  // ---- LN2 (reads d_out) ----
  ln_f16_kernel<<<TM, 256, 0, stream>>>(out, g2, be2, h1);

  // ---- MLP1: [4096,1024] x [1024,4096], ReLU, f16 ----
  const int blocksMLP1 = (TM / 32) * (4 * TC / 64) / 8;   // 1024 blocks
  gemm_wmma_kernel<true,  true,  true, false><<<blocksMLP1, 256, 0, stream>>>(
      h1, W1T, b1, nullptr, ff1, TM, 4 * TC, TC);

  // ---- MLP2: [4096,4096] x [4096,1024] + bias + residual(d_out) -> d_out ----
  gemm_wmma_kernel<false, false, true, true><<<blocksCxC, 256, 0, stream>>>(
      ff1, W2T, b2, out, out, TM, TC, 4 * TC);
}
